// PoHBlock_18889266167870
// MI455X (gfx1250) — matrix operand order repaired and checked by reading, not compiled, not run on hardware
//
#include <hip/hip_runtime.h>

typedef unsigned short u16;
typedef __attribute__((ext_vector_type(16))) __bf16          v16bf;
typedef __attribute__((ext_vector_type(16))) unsigned short  us16;
typedef __attribute__((ext_vector_type(8)))  float           v8f;

#define D_MODEL 1024
#define N_HEADS 16
#define D_HEAD  64
#define D_FF    4096
#define BATCH   8
#define SEQ     1024
#define MROWS   (BATCH*SEQ)      /* 8192 */
#define LDS_STR 40               /* LDS row stride (u16 elems), keeps 16B alignment */

__device__ __forceinline__ u16 f2bf(float f) {
  unsigned int u = __float_as_uint(f);
  u += 0x7FFFu + ((u >> 16) & 1u);          // round-to-nearest-even
  return (u16)(u >> 16);
}
__device__ __forceinline__ v16bf as_bf16(us16 u) { return __builtin_bit_cast(v16bf, u); }
__device__ __forceinline__ v8f wmma_bf16(v16bf a, v16bf b, v8f c) {
  return __builtin_amdgcn_wmma_f32_16x16x32_bf16(false, a, false, b, (short)0, c, false, false);
}
__device__ __forceinline__ v8f v8f_zero() {
  v8f v = {0.f,0.f,0.f,0.f,0.f,0.f,0.f,0.f};
  return v;
}

// ---- CDNA5 async global->LDS copies (ASYNCcnt-tracked, bypass VGPRs) ----
__device__ __forceinline__ unsigned lds_off32(const void* p) {
  return (unsigned)(size_t)p;               // generic addr low 32 bits = LDS offset
}
__device__ __forceinline__ void async_copy_b128(unsigned lds, const void* g) {
  asm volatile("global_load_async_to_lds_b128 %0, %1, off"
               :: "v"(lds), "v"(g) : "memory");
}
__device__ __forceinline__ void async_copy_b128_x2(unsigned lds, const void* g) {
  // INST_OFFSET is added to BOTH the LDS and global address (ISA 10. Async)
  asm volatile("global_load_async_to_lds_b128 %0, %1, off\n\t"
               "global_load_async_to_lds_b128 %0, %1, off offset:16"
               :: "v"(lds), "v"(g) : "memory");
}
__device__ __forceinline__ void wait_async0() {
  asm volatile("s_wait_asynccnt 0" ::: "memory");
}

// ---------------------------------------------------------------------------
// Weight preparation
// ---------------------------------------------------------------------------
__global__ void pack_qkv_w(const float* __restrict__ Wq, const float* __restrict__ Wk,
                           const float* __restrict__ Wv, u16* __restrict__ out) {
  int idx = blockIdx.x * 256 + threadIdx.x;               // over 1024*3072
  if (idx >= D_MODEL * 3 * D_MODEL) return;
  int d = idx / (3 * D_MODEL);
  int n = idx % (3 * D_MODEL);
  int sec = n >> 10, c = n & 1023, h = c >> 6, k = c & 63;
  const float* W = (sec == 0) ? Wq : (sec == 1) ? Wk : Wv;   // [H, D, Dh]
  out[idx] = f2bf(W[((h << 10) + d) * D_HEAD + k]);
}

__global__ void pack_qkv_b(const float* __restrict__ bq, const float* __restrict__ bk,
                           const float* __restrict__ bv, float* __restrict__ out) {
  int idx = blockIdx.x * 256 + threadIdx.x;               // over 3072
  if (idx >= 3 * D_MODEL) return;
  int sec = idx >> 10, c = idx & 1023;
  out[idx] = (sec == 0) ? bq[c] : (sec == 1) ? bk[c] : bv[c];
}

__global__ void conv_bf16(const float* __restrict__ in, u16* __restrict__ out, int n) {
  int idx = blockIdx.x * 256 + threadIdx.x;
  if (idx < n) out[idx] = f2bf(in[idx]);
}

// ---------------------------------------------------------------------------
// Tiled WMMA GEMM: C[M,N] = act(A[M,K] @ B[K,N] + bias)
// Block tile 128x128, K-step 32, 8 waves (2x4), wave tile 64x32.
// A: fp32 (VGPR-convert path) or bf16 (async-to-LDS path).  B: bf16 [K,N].
// ---------------------------------------------------------------------------
template<bool A_F32, bool RELU, bool WF32, bool WBF16>
__global__ __launch_bounds__(256)
void gemm_bf16_kernel(const void* __restrict__ Aop, const u16* __restrict__ Bw,
                      const float* __restrict__ bias,
                      float* __restrict__ Cf, u16* __restrict__ Ch,
                      int M, int N, int K) {
  __shared__ __attribute__((aligned(16))) u16 As[128 * LDS_STR];  // [row][k]
  __shared__ __attribute__((aligned(16))) u16 Bs[128 * LDS_STR];  // [n][k] transposed

  const int tid   = threadIdx.x;
  const int lane  = tid & 31;
  const int wave  = tid >> 5;
  const int waveM = wave >> 2;           // 0..1 -> 64 rows
  const int waveN = wave & 3;            // 0..3 -> 32 cols
  const long mBlock = (long)blockIdx.y * 128;
  const long nBlock = (long)blockIdx.x * 128;

  v8f acc[4][2];
  #pragma unroll
  for (int i = 0; i < 4; i++)
    #pragma unroll
    for (int j = 0; j < 2; j++) acc[i][j] = v8f_zero();

  const int arow  = tid >> 1;            // 0..127
  const int akoff = (tid & 1) * 16;      // 0 | 16
  const int bkrow = tid >> 3;            // 0..31
  const int bnoff = (tid & 7) * 16;      // 0..112

  const int lr    = lane & 15;
  const int kb    = (lane >> 4) * 8;     // A-fragment K base per ISA layout
  const int khalf = (lane >> 4) * 16;    // unused; B fragment also uses kb

  const unsigned asA = lds_off32(&As[arow * LDS_STR + akoff]);

  for (int k0 = 0; k0 < K; k0 += 32) {
    __syncthreads();
    // ---- stage A tile ----
    if (A_F32) {
      const float* A = ((const float*)Aop) + (size_t)(mBlock + arow) * K + k0 + akoff;
      __builtin_prefetch(A + 32, 0, 0);
      u16* dst = &As[arow * LDS_STR + akoff];
      #pragma unroll
      for (int i = 0; i < 16; i++) dst[i] = f2bf(A[i]);
    } else {
      // bf16 A: async copy straight into LDS (two b128 = 32B per thread)
      const u16* A = ((const u16*)Aop) + (size_t)(mBlock + arow) * K + k0 + akoff;
      __builtin_prefetch(A + 32, 0, 0);
      async_copy_b128_x2(asA, A);
    }
    // ---- stage B tile transposed: Bs[n][k] (scatter, VGPR path) ----
    {
      const u16* Bp = Bw + (size_t)(k0 + bkrow) * N + nBlock + bnoff;
      __builtin_prefetch(Bp + (size_t)32 * N, 0, 0);
      #pragma unroll
      for (int i = 0; i < 16; i++) Bs[(bnoff + i) * LDS_STR + bkrow] = Bp[i];
    }
    if (!A_F32) wait_async0();
    __syncthreads();

    // ---- load fragments ----
    v16bf aF[4], bF[2];
    #pragma unroll
    for (int mt = 0; mt < 4; mt++) {
      const u16* ap = &As[(waveM * 64 + mt * 16 + lr) * LDS_STR + kb];
      us16 u;
      #pragma unroll
      for (int j = 0; j < 8; j++) { u[j] = ap[j]; u[8 + j] = ap[16 + j]; }
      aF[mt] = as_bf16(u);
    }
    #pragma unroll
    for (int nt = 0; nt < 2; nt++) {
      const u16* bp = &Bs[(waveN * 32 + nt * 16 + lr) * LDS_STR + kb];
      us16 u;
      #pragma unroll
      for (int j = 0; j < 8; j++) { u[j] = bp[j]; u[8 + j] = bp[16 + j]; }
      bF[nt] = as_bf16(u);
    }
    // ---- 8 WMMAs ----
    #pragma unroll
    for (int mt = 0; mt < 4; mt++)
      #pragma unroll
      for (int nt = 0; nt < 2; nt++)
        acc[mt][nt] = wmma_bf16(aF[mt], bF[nt], acc[mt][nt]);
  }

  // ---- epilogue: bias (+ReLU), write fp32 and/or bf16 ----
  const int rbase = (lane >> 4) * 8;     // C layout: lanes 16-31 hold rows 8..15
  #pragma unroll
  for (int mt = 0; mt < 4; mt++) {
    #pragma unroll
    for (int nt = 0; nt < 2; nt++) {
      const long col = nBlock + waveN * 32 + nt * 16 + lr;
      const float bv = bias[col];
      #pragma unroll
      for (int i = 0; i < 8; i++) {
        const long row = mBlock + waveM * 64 + mt * 16 + rbase + i;
        float v = acc[mt][nt][i] + bv;
        if (RELU) v = fmaxf(v, 0.0f);
        if (WF32)  Cf[(size_t)row * N + col] = v;
        if (WBF16) Ch[(size_t)row * N + col] = f2bf(v);
      }
    }
  }
}

// ---------------------------------------------------------------------------
// Flash attention.  qkv: [B*T, 3072] bf16 (q | k | v sections, per-head 64 cols)
// grid (B*H, T/128), 8 waves/block, one wave per 16-query tile, 32-key blocks.
// K tile staged via async-to-LDS; V staged transposed via VGPR path.
// ---------------------------------------------------------------------------
__global__ __launch_bounds__(256)
void attn_kernel(const u16* __restrict__ qkv, u16* __restrict__ attn_o) {
  __shared__ __attribute__((aligned(16))) u16 Ks[32 * 72];          // [key][d]
  __shared__ __attribute__((aligned(16))) u16 Vts[64 * LDS_STR];    // [d][key]
  __shared__ __attribute__((aligned(16))) u16 Ps[8 * 16 * LDS_STR]; // per-wave P

  const int tid  = threadIdx.x, lane = tid & 31, wave = tid >> 5;
  const int b    = blockIdx.x >> 4;
  const int h    = blockIdx.x & 15;
  const int q0   = blockIdx.y * 128 + wave * 16;
  const size_t rowBase = (size_t)b * SEQ;
  const int lr    = lane & 15;
  const int kb    = (lane >> 4) * 8;
  const int khalf = (lane >> 4) * 16;
  const int rbase = (lane >> 4) * 8;

  // Q fragments: A layout, 16 queries x (two 32-wide d windows)
  v16bf qF[2];
  {
    const u16* qp = qkv + (rowBase + q0 + lr) * (size_t)(3 * D_MODEL) + h * D_HEAD;
    #pragma unroll
    for (int w = 0; w < 2; w++) {
      us16 u;
      #pragma unroll
      for (int j = 0; j < 8; j++) {
        u[j]     = qp[w * 32 + kb + j];
        u[8 + j] = qp[w * 32 + 16 + kb + j];
      }
      qF[w] = as_bf16(u);
    }
  }

  v8f o[4];
  #pragma unroll
  for (int t = 0; t < 4; t++) o[t] = v8f_zero();
  float m[8], l[8];
  #pragma unroll
  for (int i = 0; i < 8; i++) { m[i] = -1e30f; l[i] = 0.0f; }
  const float scale = 0.125f;            // 1/sqrt(64)

  const int stg_key = tid >> 3;          // 0..31
  const int stg_d   = (tid & 7) * 8;     // 0..56
  const unsigned ldsK = lds_off32(&Ks[stg_key * 72 + stg_d]);

  for (int kb0 = 0; kb0 < SEQ; kb0 += 32) {
    __syncthreads();
    // ---- stage K (async, natural) and V (transposed, VGPR) for this head ----
    {
      const size_t r = (rowBase + kb0 + stg_key) * (size_t)(3 * D_MODEL);
      const u16* kp = qkv + r + D_MODEL     + h * D_HEAD + stg_d;
      const u16* vp = qkv + r + 2 * D_MODEL + h * D_HEAD + stg_d;
      async_copy_b128(ldsK, kp);                 // 16B/lane straight to LDS
      #pragma unroll
      for (int i = 0; i < 8; i++) Vts[(stg_d + i) * LDS_STR + stg_key] = vp[i];
    }
    wait_async0();
    __syncthreads();

    // ---- S = Q . K^T  (two 16x16 tiles over 32 keys; chain over d) ----
    v8f s0 = v8f_zero(), s1 = v8f_zero();
    #pragma unroll
    for (int w = 0; w < 2; w++) {
      us16 u0, u1;
      const u16* k0p = &Ks[lr * 72        + w * 32 + kb];
      const u16* k1p = &Ks[(16 + lr) * 72 + w * 32 + kb];
      #pragma unroll
      for (int j = 0; j < 8; j++) { u0[j] = k0p[j]; u1[j] = k1p[j]; u0[8 + j] = k0p[16 + j]; u1[8 + j] = k1p[16 + j]; }
      s0 = wmma_bf16(qF[w], as_bf16(u0), s0);
      s1 = wmma_bf16(qF[w], as_bf16(u1), s1);
    }

    // ---- online softmax update (row stats across 16-lane halves) ----
    float p0[8], p1[8], cf[8];
    #pragma unroll
    for (int i = 0; i < 8; i++) {
      float a = s0[i] * scale, c = s1[i] * scale;
      float mx = fmaxf(a, c);
      #pragma unroll
      for (int off = 1; off < 16; off <<= 1) mx = fmaxf(mx, __shfl_xor(mx, off));
      float mn = fmaxf(m[i], mx);
      cf[i] = __expf(m[i] - mn);
      m[i]  = mn;
      p0[i] = __expf(a - mn);
      p1[i] = __expf(c - mn);
      float ls = p0[i] + p1[i];
      #pragma unroll
      for (int off = 1; off < 16; off <<= 1) ls += __shfl_xor(ls, off);
      l[i] = l[i] * cf[i] + ls;
    }
    #pragma unroll
    for (int t = 0; t < 4; t++)
      #pragma unroll
      for (int i = 0; i < 8; i++) o[t][i] *= cf[i];

    // ---- relayout P (C layout -> A fragment) through per-wave LDS ----
    u16* pw = &Ps[wave * 16 * LDS_STR];
    #pragma unroll
    for (int i = 0; i < 8; i++) {
      const int r = rbase + i;
      pw[r * LDS_STR + lr]      = f2bf(p0[i]);
      pw[r * LDS_STR + 16 + lr] = f2bf(p1[i]);
    }
    __syncthreads();

    us16 up;
    const u16* pr = &Ps[wave * 16 * LDS_STR + lr * LDS_STR + kb];
    #pragma unroll
    for (int j = 0; j < 8; j++) { up[j] = pr[j]; up[8 + j] = pr[16 + j]; }
    v16bf pF = as_bf16(up);

    // ---- O += P . V ----
    #pragma unroll
    for (int dt = 0; dt < 4; dt++) {
      us16 uv;
      const u16* vr = &Vts[(dt * 16 + lr) * LDS_STR + kb];
      #pragma unroll
      for (int j = 0; j < 8; j++) { uv[j] = vr[j]; uv[8 + j] = vr[16 + j]; }
      o[dt] = wmma_bf16(pF, as_bf16(uv), o[dt]);
    }
  }

  // ---- finalize: O / l, write bf16 [row, h*64 + d] ----
  float inv[8];
  #pragma unroll
  for (int i = 0; i < 8; i++) inv[i] = 1.0f / l[i];
  #pragma unroll
  for (int dt = 0; dt < 4; dt++)
    #pragma unroll
    for (int i = 0; i < 8; i++) {
      const int row = q0 + rbase + i;
      const int col = h * D_HEAD + dt * 16 + lr;
      attn_o[(rowBase + row) * (size_t)D_MODEL + col] = f2bf(o[dt][i] * inv[i]);
    }
}

// ---------------------------------------------------------------------------
// Residual + LayerNorm: out = LN(zres + x) * g + b   (one row per block)
// ---------------------------------------------------------------------------
__global__ __launch_bounds__(256)
void ln_kernel(const float* __restrict__ zres, const float* __restrict__ x,
               const float* __restrict__ g, const float* __restrict__ beta,
               float* __restrict__ out) {
  __shared__ float sred[2][8];
  const int row = blockIdx.x, t = threadIdx.x;
  const float* zr = zres + (size_t)row * D_MODEL;
  const float* xr = x    + (size_t)row * D_MODEL;

  float s[4], sum = 0.f, sq = 0.f;
  #pragma unroll
  for (int i = 0; i < 4; i++) {
    const int idx = t + i * 256;
    const float v = zr[idx] + xr[idx];
    s[i] = v; sum += v; sq += v * v;
  }
  #pragma unroll
  for (int off = 16; off > 0; off >>= 1) {
    sum += __shfl_xor(sum, off);
    sq  += __shfl_xor(sq, off);
  }
  if ((t & 31) == 0) { sred[0][t >> 5] = sum; sred[1][t >> 5] = sq; }
  __syncthreads();
  sum = 0.f; sq = 0.f;
  #pragma unroll
  for (int w = 0; w < 8; w++) { sum += sred[0][w]; sq += sred[1][w]; }

  const float mean = sum * (1.0f / D_MODEL);
  const float var  = sq  * (1.0f / D_MODEL) - mean * mean;
  const float rstd = rsqrtf(var + 1e-5f);
  #pragma unroll
  for (int i = 0; i < 4; i++) {
    const int idx = t + i * 256;
    out[(size_t)row * D_MODEL + idx] = (s[i] - mean) * rstd * g[idx] + beta[idx];
  }
}

// ---------------------------------------------------------------------------
// Orchestration
// ---------------------------------------------------------------------------
extern "C" void kernel_launch(void* const* d_in, const int* in_sizes, int n_in,
                              void* d_out, int out_size, void* d_ws, size_t ws_size,
                              hipStream_t stream) {
  (void)in_sizes; (void)n_in; (void)out_size; (void)ws_size;
  const float* z   = (const float*)d_in[0];
  const float* Wq  = (const float*)d_in[1];
  const float* bq  = (const float*)d_in[2];
  const float* Wk  = (const float*)d_in[3];
  const float* bk  = (const float*)d_in[4];
  const float* Wv  = (const float*)d_in[5];
  const float* bv  = (const float*)d_in[6];
  const float* Wo  = (const float*)d_in[7];
  const float* bo  = (const float*)d_in[8];
  const float* W1  = (const float*)d_in[9];
  const float* b1  = (const float*)d_in[10];
  const float* W2  = (const float*)d_in[11];
  const float* b2  = (const float*)d_in[12];
  const float* g1  = (const float*)d_in[13];
  const float* be1 = (const float*)d_in[14];
  const float* g2  = (const float*)d_in[15];
  const float* be2 = (const float*)d_in[16];

  char* ws = (char*)d_ws;
  u16*   Wqkv_h = (u16*)ws;              ws += (size_t)D_MODEL * 3 * D_MODEL * 2;
  u16*   Wo_h   = (u16*)ws;              ws += (size_t)D_MODEL * D_MODEL * 2;
  u16*   W1_h   = (u16*)ws;              ws += (size_t)D_MODEL * D_FF * 2;
  u16*   W2_h   = (u16*)ws;              ws += (size_t)D_FF * D_MODEL * 2;
  float* bqkv   = (float*)ws;            ws += (size_t)3 * D_MODEL * 4;
  u16*   qkv    = (u16*)ws;              ws += (size_t)MROWS * 3 * D_MODEL * 2;
  u16*   attn_o = (u16*)ws;              ws += (size_t)MROWS * D_MODEL * 2;
  float* xbuf   = (float*)ws;            ws += (size_t)MROWS * D_MODEL * 4;
  u16*   h1     = (u16*)ws;              ws += (size_t)MROWS * D_FF * 2;
  float* zB     = (float*)ws;            ws += (size_t)MROWS * D_MODEL * 4;
  float* zC     = (float*)ws;            ws += (size_t)MROWS * D_MODEL * 4;

  // --- weight prep (bf16) ---
  pack_qkv_w<<<(D_MODEL * 3 * D_MODEL + 255) / 256, 256, 0, stream>>>(Wq, Wk, Wv, Wqkv_h);
  pack_qkv_b<<<(3 * D_MODEL + 255) / 256, 256, 0, stream>>>(bq, bk, bv, bqkv);
  conv_bf16<<<(D_MODEL * D_MODEL + 255) / 256, 256, 0, stream>>>(Wo, Wo_h, D_MODEL * D_MODEL);
  conv_bf16<<<(D_MODEL * D_FF + 255) / 256, 256, 0, stream>>>(W1, W1_h, D_MODEL * D_FF);
  conv_bf16<<<(D_FF * D_MODEL + 255) / 256, 256, 0, stream>>>(W2, W2_h, D_FF * D_MODEL);

  const float* zcur = z;
  for (int it = 0; it < 2; ++it) {
    // fused QKV projection: [8192,1024] @ [1024,3072] -> bf16
    gemm_bf16_kernel<true, false, false, true>
        <<<dim3(3 * D_MODEL / 128, MROWS / 128), 256, 0, stream>>>(
            zcur, Wqkv_h, bqkv, nullptr, qkv, MROWS, 3 * D_MODEL, D_MODEL);
    // flash attention -> attn_o bf16 [8192,1024]
    attn_kernel<<<dim3(BATCH * N_HEADS, SEQ / 128), 256, 0, stream>>>(qkv, attn_o);
    // output projection -> xbuf fp32
    gemm_bf16_kernel<false, false, true, false>
        <<<dim3(D_MODEL / 128, MROWS / 128), 256, 0, stream>>>(
            attn_o, Wo_h, bo, xbuf, nullptr, MROWS, D_MODEL, D_MODEL);
    // z = LN(z + attn_out); iteration 2's LN1 output IS z_refined -> d_out
    float* ln1out = (it == 1) ? (float*)d_out : zB;
    ln_kernel<<<MROWS, 256, 0, stream>>>(zcur, xbuf, g1, be1, ln1out);

    if (it == 0) {                       // final iteration's FFN never affects z_refined
      gemm_bf16_kernel<true, true, false, true>
          <<<dim3(D_FF / 128, MROWS / 128), 256, 0, stream>>>(
              zB, W1_h, b1, nullptr, h1, MROWS, D_FF, D_MODEL);
      gemm_bf16_kernel<false, false, true, false>
          <<<dim3(D_MODEL / 128, MROWS / 128), 256, 0, stream>>>(
              h1, W2_h, b2, xbuf, nullptr, MROWS, D_MODEL, D_FF);
      ln_kernel<<<MROWS, 256, 0, stream>>>(zB, xbuf, g2, be2, zC);
      zcur = zC;
    }
  }
}
